// MiniGINv3_58188216926530
// MI455X (gfx1250) — compile-verified, matrix operand order checked by reading, not run on hardware
//
#include <hip/hip_runtime.h>

typedef __bf16 bf16;
typedef __attribute__((ext_vector_type(16))) __bf16 v16bf;
typedef __attribute__((ext_vector_type(8)))  float  v8f;
typedef __attribute__((ext_vector_type(4)))  int    v4i;

// ---- CDNA5 async global->LDS copy (gfx1250), guarded for toolchain support ----
#if defined(__has_builtin)
#if __has_builtin(__builtin_amdgcn_global_load_async_to_lds_b128) && \
    __has_builtin(__builtin_amdgcn_s_wait_asynccnt)
#define USE_ASYNC_LDS 1
#endif
#endif

namespace {

constexpr int kN  = 50000;
constexpr int kE  = 400000;
constexpr int kB  = 128;
constexpr int kD  = 774;
constexpr int kDP = 800;   // D padded to multiple of 32 (WMMA bf16 K-step)
constexpr int kH  = 384;
constexpr int k2H = 768;

constexpr int TM = 128;
constexpr int TN = 128;
constexpr int TK = 32;
constexpr int LDSW = TK + 8;   // bf16 elems per LDS row; pad to dodge bank conflicts

#ifdef USE_ASYNC_LDS
typedef __attribute__((address_space(1))) v4i* gv4i_p;
typedef __attribute__((address_space(3))) v4i* lv4i_p;

__device__ __forceinline__ void async_cp16(const bf16* gsrc, bf16* ldst) {
  // 16-byte async copy, tracked by ASYNCcnt; no VGPR data movement
  v4i* g = (v4i*)gsrc;   // generic, const dropped
  v4i* l = (v4i*)ldst;   // generic
  __builtin_amdgcn_global_load_async_to_lds_b128((gv4i_p)g, (lv4i_p)l, 0, 0);
}
#endif

// ------------------------------------------------------------------
// pack f32 -> bf16, padding the column (K) dimension with zeros (activations)
__global__ void pack_pad_cols(const float* __restrict__ src, bf16* __restrict__ dst,
                              int rows, int cols, int cols_pad) {
  long long i = blockIdx.x * (long long)blockDim.x + threadIdx.x;
  long long total = (long long)rows * cols_pad;
  if (i >= total) return;
  int r = (int)(i / cols_pad);
  int c = (int)(i % cols_pad);
  dst[i] = (c < cols) ? (bf16)src[(long long)r * cols + c] : (bf16)0.0f;
}

// pack weight f32 [K, N] -> bf16 TRANSPOSED [N, Kpad] (zero-pad K).
// B tiles then have the same contiguous-K layout as A tiles -> async b128 copies.
__global__ void pack_wT(const float* __restrict__ src, bf16* __restrict__ dst,
                        int K, int Ncols, int Kpad) {
  long long i = blockIdx.x * (long long)blockDim.x + threadIdx.x;
  long long total = (long long)Ncols * Kpad;
  if (i >= total) return;
  int n = (int)(i / Kpad);
  int k = (int)(i % Kpad);
  dst[i] = (k < K) ? (bf16)src[(long long)k * Ncols + n] : (bf16)0.0f;
}

// ------------------------------------------------------------------
// C[M,Nn] = A[M,K](bf16) @ Bt[Nn,K](bf16, pre-transposed) + bias, f32 out.
// 256 threads = 8 wave32; block tile 128x128, K-step 32, double-buffered LDS.
__global__ __launch_bounds__(256)
void gemm_bf16_wmma(const bf16* __restrict__ A, const bf16* __restrict__ Bt,
                    const float* __restrict__ bias, float* __restrict__ C,
                    int M, int Nn, int K) {
  __shared__ bf16 As[2][TM * LDSW];
  __shared__ bf16 Bs[2][TN * LDSW];

  const int tid  = threadIdx.x;
  const int wave = tid >> 5;
  const int lane = tid & 31;
  const int m0 = blockIdx.y * TM;
  const int n0 = blockIdx.x * TN;

  v8f acc[8];
#pragma unroll
  for (int g = 0; g < 8; ++g)
#pragma unroll
    for (int j = 0; j < 8; ++j) acc[g][j] = 0.0f;

  // ---- copy roles: thread -> (row, 16-elem K chunk); A and B tiles same shape
  const int crow = tid >> 1;          // 0..127
  const int ckk  = (tid & 1) * 16;    // 0 / 16
  int arow = m0 + crow;
  if (arow >= M) arow = M - 1;        // clamp: padded rows masked at store
  const bf16* agp = A  + (long long)arow * K + ckk;
  const bf16* bgp = Bt + (long long)(n0 + crow) * K + ckk;
  bf16* alp = &As[0][crow * LDSW + ckk];
  bf16* blp = &Bs[0][crow * LDSW + ckk];

  const int ntiles = K / TK;

  // ---- fragment roles
  const int frow = lane & 15;            // M (A) / N (B) within 16
  const int ku   = (lane >> 4) << 2;     // uint idx of K-half base (0 or 4)

  auto compute = [&](int b) {
    union { unsigned int u[8]; v16bf v; } af;
    const unsigned int* ap = (const unsigned int*)&As[b][(16 * wave + frow) * LDSW];
#pragma unroll
    for (int j = 0; j < 4; ++j) {
      af.u[j]     = ap[ku + j];
      af.u[4 + j] = ap[ku + 8 + j];
    }
#pragma unroll
    for (int g = 0; g < 8; ++g) {
      union { unsigned int u[8]; v16bf v; } bfrag;
      const unsigned int* bp = (const unsigned int*)&Bs[b][(16 * g + frow) * LDSW];
#pragma unroll
      for (int j = 0; j < 4; ++j) {
        bfrag.u[j]     = bp[ku + j];
        bfrag.u[4 + j] = bp[ku + 8 + j];
      }
      acc[g] = __builtin_amdgcn_wmma_f32_16x16x32_bf16(
          false, af.v, false, bfrag.v, (short)0, acc[g], false, false);
    }
  };

#ifdef USE_ASYNC_LDS
  auto issue = [&](int t) {
    const bf16* ag = agp + t * TK;
    const bf16* bg = bgp + t * TK;
    bf16* al = alp + (t & 1) * (TM * LDSW);
    bf16* bl = blp + (t & 1) * (TN * LDSW);
    async_cp16(ag, al);
    async_cp16(ag + 8, al + 8);
    async_cp16(bg, bl);
    async_cp16(bg + 8, bl + 8);
  };
  issue(0);
  for (int t = 0; t < ntiles; ++t) {
    if (t + 1 < ntiles) {
      issue(t + 1);
      // async loads complete in order: <=4 outstanding => tile t has landed
      __builtin_amdgcn_s_wait_asynccnt(4);
    } else {
      __builtin_amdgcn_s_wait_asynccnt(0);
    }
    __syncthreads();   // tile t visible to all waves; prev buffer free for t+1
    compute(t & 1);
    __syncthreads();   // all waves done reading before buffer is refilled
  }
#else
  for (int t = 0; t < ntiles; ++t) {
    const uint4* asrc = (const uint4*)(agp + t * TK);
    const uint4* bsrc = (const uint4*)(bgp + t * TK);
    uint4 a0 = asrc[0], a1 = asrc[1];
    uint4 b0 = bsrc[0], b1 = bsrc[1];
    uint4* ad = (uint4*)(alp + (t & 1) * (TM * LDSW));
    uint4* bd = (uint4*)(blp + (t & 1) * (TN * LDSW));
    ad[0] = a0; ad[1] = a1;
    bd[0] = b0; bd[1] = b1;
    __syncthreads();
    compute(t & 1);
    __syncthreads();
  }
#endif

  // ---- epilogue: acc elem j -> row 16w + j + 8*(lane>=16), col 16g + (lane&15)
#pragma unroll
  for (int g = 0; g < 8; ++g) {
    int col = n0 + 16 * g + (lane & 15);
    float bv = bias ? bias[col] : 0.0f;
#pragma unroll
    for (int j = 0; j < 8; ++j) {
      int row = m0 + 16 * wave + j + ((lane >> 4) << 3);
      if (row < M) C[(long long)row * Nn + col] = acc[g][j] + bv;
    }
  }
}

// ------------------------------------------------------------------
// column sums / sum of squares for batch-norm (phase 1)
__global__ void col_stats(const float* __restrict__ z, float* __restrict__ sums,
                          int M, int Nn) {
  const int ROWS = 512;
  int c = blockIdx.y * 128 + threadIdx.x;
  int r0 = blockIdx.x * ROWS;
  int r1 = r0 + ROWS;
  if (r1 > M) r1 = M;
  float s = 0.0f, s2 = 0.0f;
  for (int r = r0; r < r1; ++r) {
    float v = z[(long long)r * Nn + c];
    s += v;
    s2 += v * v;
  }
  atomicAdd(&sums[c], s);
  atomicAdd(&sums[Nn + c], s2);
}

// phase 2: scale = g * rsqrt(var+eps); shift = b - mean*scale
__global__ void bn_finalize(const float* __restrict__ sums, const float* __restrict__ g,
                            const float* __restrict__ b, float* __restrict__ scsh,
                            int Nn, float invM) {
  int c = blockIdx.x * blockDim.x + threadIdx.x;
  if (c >= Nn) return;
  float m = sums[c] * invM;
  float v = sums[Nn + c] * invM - m * m;
  float a = g[c] * rsqrtf(v + 1e-5f);
  scsh[c] = a;
  scsh[Nn + c] = b[c] - m * a;
}

// phase 3: y = relu(z*scale + shift) [+ residual], write f32 and/or bf16
template <bool RESID, bool WF32, bool WBF16>
__global__ void bn_relu_apply(const float* __restrict__ z, const float* __restrict__ scsh,
                              float* __restrict__ resid_io, float* __restrict__ out_f32,
                              bf16* __restrict__ out_bf16, int M, int Nn) {
  long long i = blockIdx.x * (long long)blockDim.x + threadIdx.x;
  long long total = (long long)M * Nn;
  if (i >= total) return;
  int c = (int)(i % Nn);
  float y = fmaxf(fmaf(z[i], scsh[c], scsh[Nn + c]), 0.0f);
  if (RESID) {
    y += resid_io[i];
    resid_io[i] = y;
  }
  if (WF32) out_f32[i] = y;
  if (WBF16) out_bf16[i] = (bf16)y;
}

// ------------------------------------------------------------------
// GIN aggregation: agg[dst] += h[src], 4 features per thread
__global__ void edge_scatter_add(const int* __restrict__ ei, const float* __restrict__ h,
                                 float* __restrict__ agg, int E, int H) {
  long long i = blockIdx.x * (long long)blockDim.x + threadIdx.x;
  int chunks = H >> 2;
  long long total = (long long)E * chunks;
  if (i >= total) return;
  int e = (int)(i / chunks);
  int c4 = (int)(i % chunks) << 2;
  int s = ei[e];       // edge_index[0][e]
  int d = ei[E + e];   // edge_index[1][e]
  const float4 hv = *(const float4*)&h[(long long)s * H + c4];
  float* ag = &agg[(long long)d * H + c4];
  atomicAdd(ag + 0, hv.x);
  atomicAdd(ag + 1, hv.y);
  atomicAdd(ag + 2, hv.z);
  atomicAdd(ag + 3, hv.w);
}

// z = (1+eps)*h + agg, packed to bf16 as next GEMM's A operand
__global__ void gin_combine(const float* __restrict__ h, const float* __restrict__ agg,
                            const float* __restrict__ epsp, bf16* __restrict__ out,
                            long long total) {
  long long i = blockIdx.x * (long long)blockDim.x + threadIdx.x;
  if (i >= total) return;
  float e = 1.0f + epsp[0];
  out[i] = (bf16)(fmaf(e, h[i], agg[i]));
}

// ------------------------------------------------------------------
// pooling: addpool[b] += h[i]; counts[b] += 1
__global__ void pool_scatter(const float* __restrict__ h, const int* __restrict__ batch,
                             float* __restrict__ addpool, float* __restrict__ counts,
                             int N, int H) {
  long long i = blockIdx.x * (long long)blockDim.x + threadIdx.x;
  int chunks = H >> 2;
  long long total = (long long)N * chunks;
  if (i >= total) return;
  int node = (int)(i / chunks);
  int c4 = (int)(i % chunks) << 2;
  int b = batch[node];
  const float4 hv = *(const float4*)&h[(long long)node * H + c4];
  float* ap = &addpool[(long long)b * H + c4];
  atomicAdd(ap + 0, hv.x);
  atomicAdd(ap + 1, hv.y);
  atomicAdd(ap + 2, hv.z);
  atomicAdd(ap + 3, hv.w);
  if (c4 == 0) atomicAdd(&counts[b], 1.0f);
}

// emb = concat(mean_pool, add_pool)
__global__ void pool_finalize(const float* __restrict__ addpool, const float* __restrict__ counts,
                              float* __restrict__ emb, int B, int H) {
  int i = blockIdx.x * blockDim.x + threadIdx.x;
  if (i >= B * H) return;
  int b = i / H, c = i % H;
  float a = addpool[i];
  float cnt = fmaxf(counts[b], 1.0f);
  emb[(long long)b * 2 * H + c] = a / cnt;
  emb[(long long)b * 2 * H + H + c] = a;
}

// ------------------------------------------------------------------
// tiny FC: Y[b,o] = act(X[b,:] . W[:,o] + bias[o]);  ACT: 0 none, 1 relu, 2 sigmoid
template <int ACT>
__global__ void fc_kernel(const float* __restrict__ X, const float* __restrict__ W,
                          const float* __restrict__ bias, float* __restrict__ Y,
                          int B, int In, int Out) {
  int i = blockIdx.x * blockDim.x + threadIdx.x;
  if (i >= B * Out) return;
  int b = i / Out, o = i % Out;
  float s = bias[o];
  for (int k = 0; k < In; ++k) s = fmaf(X[(long long)b * In + k], W[(long long)k * Out + o], s);
  if (ACT == 1) s = fmaxf(s, 0.0f);
  if (ACT == 2) s = 1.0f / (1.0f + __expf(-s));
  Y[i] = s;
}

inline unsigned int ceil_div_ll(long long a, long long b) {
  return (unsigned int)((a + b - 1) / b);
}

}  // namespace

extern "C" void kernel_launch(void* const* d_in, const int* in_sizes, int n_in,
                              void* d_out, int out_size, void* d_ws, size_t ws_size,
                              hipStream_t stream) {
  (void)in_sizes; (void)n_in; (void)out_size; (void)ws_size;

  // ---- inputs (jax pytree flatten: dict keys in sorted order) ----
  const float* x     = (const float*)d_in[0];
  const int*   ei    = (const int*)d_in[1];
  const int*   batch = (const int*)d_in[2];
  const float* Wc1 = (const float*)d_in[3];
  const float* Wc2 = (const float*)d_in[4];
  const float* Wcf = (const float*)d_in[5];
  const float* Win = (const float*)d_in[6];
  const float* b_in = (const float*)d_in[7];
  const float* bc1 = (const float*)d_in[8];
  const float* bc2 = (const float*)d_in[9];
  const float* bcf = (const float*)d_in[10];
  const float* bin = (const float*)d_in[11];
  const float* g_in = (const float*)d_in[12];
  struct Layer { const float *W1, *W2, *b, *b1, *b2, *b_mid, *eps, *g, *g_mid; };
  Layer L[3];
  for (int l = 0; l < 3; ++l) {
    int base = 13 + 9 * l;  // sorted: W1,W2,b,b1,b2,b_mid,eps,g,g_mid
    L[l].W1    = (const float*)d_in[base + 0];
    L[l].W2    = (const float*)d_in[base + 1];
    L[l].b     = (const float*)d_in[base + 2];
    L[l].b1    = (const float*)d_in[base + 3];
    L[l].b2    = (const float*)d_in[base + 4];
    L[l].b_mid = (const float*)d_in[base + 5];
    L[l].eps   = (const float*)d_in[base + 6];
    L[l].g     = (const float*)d_in[base + 7];
    L[l].g_mid = (const float*)d_in[base + 8];
  }
  float* out = (float*)d_out;  // [0..255] logits 128x2, [256..383] confidence 128x1

  // ---- workspace carve ----
  char* p = (char*)d_ws;
  auto carve = [&](size_t bytes) -> char* {
    char* r = p;
    p += (bytes + 255) & ~(size_t)255;
    return r;
  };
  bf16*  xb    = (bf16*)carve((size_t)kN * kDP * 2);        // x packed bf16 [N,800]
  bf16*  WinT  = (bf16*)carve((size_t)kH * kDP * 2);        // Win^T  [384,800]
  bf16*  W1T[3]; bf16* W2T[3];
  for (int l = 0; l < 3; ++l) {
    W1T[l] = (bf16*)carve((size_t)k2H * kH * 2);            // W1^T [768,384]
    W2T[l] = (bf16*)carve((size_t)kH * k2H * 2);            // W2^T [384,768]
  }
  float* z     = (float*)carve((size_t)kN * k2H * 4);       // GEMM out (pre-BN), widest
  float* h     = (float*)carve((size_t)kN * kH * 4);        // node features f32 (residual)
  bf16*  ab    = (bf16*)carve((size_t)kN * k2H * 2);        // bf16 A operand (384 or 768 wide)
  float* agg   = (float*)carve((size_t)kN * kH * 4);        // scatter-add accumulator
  float* stats = (float*)carve((size_t)2 * k2H * 4);        // column sums / sumsq
  float* scsh  = (float*)carve((size_t)2 * k2H * 4);        // BN scale/shift
  float* apool = (float*)carve((size_t)kB * kH * 4);
  float* cnts  = (float*)carve((size_t)kB * 4);
  float* emb   = (float*)carve((size_t)kB * k2H * 4);
  float* hid   = (float*)carve((size_t)kB * kH * 4);

  const dim3 blk(256);

  // ---- pack input / weights (weights pre-transposed to [N][Kpad]) ----
  pack_pad_cols<<<ceil_div_ll((long long)kN * kDP, 256), blk, 0, stream>>>(x, xb, kN, kD, kDP);
  pack_wT<<<ceil_div_ll((long long)kH * kDP, 256), blk, 0, stream>>>(Win, WinT, kD, kH, kDP);
  for (int l = 0; l < 3; ++l) {
    pack_wT<<<ceil_div_ll((long long)k2H * kH, 256), blk, 0, stream>>>(L[l].W1, W1T[l], kH, k2H, kH);
    pack_wT<<<ceil_div_ll((long long)kH * k2H, 256), blk, 0, stream>>>(L[l].W2, W2T[l], k2H, kH, k2H);
  }

  const dim3 gH(kH / TN, (kN + TM - 1) / TM);    // 3 x 391
  const dim3 g2H(k2H / TN, (kN + TM - 1) / TM);  // 6 x 391

  // ---- input layer: h = relu(bn(x @ Win + bin)) ----
  gemm_bf16_wmma<<<gH, blk, 0, stream>>>(xb, WinT, bin, z, kN, kH, kDP);
  (void)hipMemsetAsync(stats, 0, 2 * k2H * 4, stream);
  col_stats<<<dim3((kN + 511) / 512, kH / 128), dim3(128), 0, stream>>>(z, stats, kN, kH);
  bn_finalize<<<(kH + 255) / 256, blk, 0, stream>>>(stats, g_in, b_in, scsh, kH, 1.0f / kN);
  bn_relu_apply<false, true, false><<<ceil_div_ll((long long)kN * kH, 256), blk, 0, stream>>>(
      z, scsh, nullptr, h, nullptr, kN, kH);

  // ---- 3 GIN layers ----
  for (int l = 0; l < 3; ++l) {
    // aggregation
    (void)hipMemsetAsync(agg, 0, (size_t)kN * kH * 4, stream);
    edge_scatter_add<<<ceil_div_ll((long long)kE * (kH / 4), 256), blk, 0, stream>>>(ei, h, agg, kE, kH);
    gin_combine<<<ceil_div_ll((long long)kN * kH, 256), blk, 0, stream>>>(
        h, agg, L[l].eps, ab, (long long)kN * kH);
    // MLP first linear + BN + ReLU
    gemm_bf16_wmma<<<g2H, blk, 0, stream>>>(ab, W1T[l], L[l].b1, z, kN, k2H, kH);
    (void)hipMemsetAsync(stats, 0, 2 * k2H * 4, stream);
    col_stats<<<dim3((kN + 511) / 512, k2H / 128), dim3(128), 0, stream>>>(z, stats, kN, k2H);
    bn_finalize<<<(k2H + 255) / 256, blk, 0, stream>>>(stats, L[l].g_mid, L[l].b_mid, scsh, k2H, 1.0f / kN);
    bn_relu_apply<false, false, true><<<ceil_div_ll((long long)kN * k2H, 256), blk, 0, stream>>>(
        z, scsh, nullptr, nullptr, ab, kN, k2H);
    // MLP second linear + BN + ReLU + residual (h <- relu(bn(z2)) + h)
    gemm_bf16_wmma<<<gH, blk, 0, stream>>>(ab, W2T[l], L[l].b2, z, kN, kH, k2H);
    (void)hipMemsetAsync(stats, 0, 2 * k2H * 4, stream);
    col_stats<<<dim3((kN + 511) / 512, kH / 128), dim3(128), 0, stream>>>(z, stats, kN, kH);
    bn_finalize<<<(kH + 255) / 256, blk, 0, stream>>>(stats, L[l].g, L[l].b, scsh, kH, 1.0f / kN);
    bn_relu_apply<true, false, false><<<ceil_div_ll((long long)kN * kH, 256), blk, 0, stream>>>(
        z, scsh, h, nullptr, nullptr, kN, kH);
  }

  // ---- pooling + classifier ----
  (void)hipMemsetAsync(apool, 0, (size_t)kB * kH * 4, stream);
  (void)hipMemsetAsync(cnts, 0, (size_t)kB * 4, stream);
  pool_scatter<<<ceil_div_ll((long long)kN * (kH / 4), 256), blk, 0, stream>>>(h, batch, apool, cnts, kN, kH);
  pool_finalize<<<(kB * kH + 255) / 256, blk, 0, stream>>>(apool, cnts, emb, kB, kH);

  fc_kernel<1><<<(kB * kH + 255) / 256, blk, 0, stream>>>(emb, Wc1, bc1, hid, kB, k2H, kH);
  fc_kernel<0><<<(kB * 2 + 255) / 256, blk, 0, stream>>>(hid, Wc2, bc2, out, kB, kH, 2);
  fc_kernel<2><<<(kB + 255) / 256, blk, 0, stream>>>(emb, Wcf, bcf, out + kB * 2, kB, k2H, 1);
}
